// ConvSTFT_14946486190465
// MI455X (gfx1250) — matrix operand order repaired and checked by reading, not compiled, not run on hardware
//
#include <hip/hip_runtime.h>

// ---------------- problem constants ----------------
#define WIN      400
#define HOP      100
#define NCH      514          // 2*(NFFT/2+1)
#define NFRAMES  4803
#define NBATCH   32
#define TLEN     480000
#define PAD      300          // WIN - HOP, both sides

// padded GEMM dims
#define KPAD     416          // 400 -> 13 * 32
#define MPAD     640          // 514 -> 40 * 16 (5 blocks.y * 8 waves)
#define XPLEN    481920       // per-batch padded f16 signal (last tile span fits exactly)
#define SPAN     1920         // halves staged in LDS per 16-frame tile (15*100+416 -> 1920)

typedef __attribute__((ext_vector_type(16))) _Float16 v16h;
typedef __attribute__((ext_vector_type(8)))  float    v8f;

// ---------------- prep: weight f32 -> f16, zero padded [MPAD][KPAD] ----------------
__global__ __launch_bounds__(256) void prep_weight(const float* __restrict__ w,
                                                   _Float16* __restrict__ wh) {
  int idx = blockIdx.x * 256 + threadIdx.x;      // MPAD*KPAD = 266240 exactly covered
  if (idx >= MPAD * KPAD) return;
  int m = idx / KPAD;
  int k = idx - m * KPAD;
  float v = (m < NCH && k < WIN) ? w[m * WIN + k] : 0.0f;
  wh[idx] = (_Float16)v;
}

// ---------------- prep: x f32 -> f16 with conv padding baked in ----------------
__global__ __launch_bounds__(256) void prep_x(const float* __restrict__ x,
                                              _Float16* __restrict__ xp) {
  size_t i = ((size_t)blockIdx.x * 256 + threadIdx.x) * 8;   // 8 halves per thread
  if (i >= (size_t)NBATCH * XPLEN) return;
  int b = (int)(i / XPLEN);
  int t = (int)(i - (size_t)b * XPLEN);                      // multiple of 8
  const float* xb = x + (size_t)b * TLEN;

  union { _Float16 h[8]; uint4 q; } u;
  if (t >= PAD && t + 8 <= PAD + TLEN) {
    // fast path: aligned float4 reads (offset 300*4 B is 16B-aligned)
    float4 f0 = *(const float4*)(xb + (t - PAD));
    float4 f1 = *(const float4*)(xb + (t - PAD) + 4);
    u.h[0] = (_Float16)f0.x; u.h[1] = (_Float16)f0.y;
    u.h[2] = (_Float16)f0.z; u.h[3] = (_Float16)f0.w;
    u.h[4] = (_Float16)f1.x; u.h[5] = (_Float16)f1.y;
    u.h[6] = (_Float16)f1.z; u.h[7] = (_Float16)f1.w;
  } else {
    for (int j = 0; j < 8; ++j) {
      int s = t + j - PAD;
      float v = (s >= 0 && s < TLEN) ? xb[s] : 0.0f;
      u.h[j] = (_Float16)v;
    }
  }
  *(uint4*)(xp + (size_t)b * XPLEN + t) = u.q;               // 16B-aligned store
}

// ---------------- main: batched GEMM via v_wmma_f32_16x16x32_f16 ----------------
__global__ __launch_bounds__(256) void stft_wmma(const _Float16* __restrict__ wh,
                                                 const _Float16* __restrict__ xp,
                                                 float* __restrict__ out) {
  __shared__ __align__(16) _Float16 xs[SPAN];    // 3840 B: x span for this 16-frame tile

  const int b     = blockIdx.z;
  const int fbase = blockIdx.x * 16;             // frame tile
  const int tid   = threadIdx.x;
  const int wave  = tid >> 5;
  const int lane  = tid & 31;

  // ---- stage x span to LDS with async global->LDS copy (ASYNCcnt path) ----
  // SPAN*2 = 3840 B = 240 lanes x 16 B; global base and LDS base are 16B-aligned.
  {
    if (tid < (SPAN * 2) / 16) {
      const _Float16* g =
          xp + (size_t)b * XPLEN + (size_t)fbase * HOP + (size_t)tid * 8;
      unsigned ldsoff = (unsigned)(uintptr_t)(void*)xs + (unsigned)tid * 16u;
      asm volatile("global_load_async_to_lds_b128 %0, %1, off"
                   :: "v"(ldsoff), "v"(g)
                   : "memory");
    }
    asm volatile("s_wait_asynccnt 0" ::: "memory");
  }
  __syncthreads();

  // ---- per-wave fragment base pointers ----
  const int mbase = (blockIdx.y * 8 + wave) * 16;          // channel tile (16 rows)
  // A fragment, this code's contiguous k order (matches B): lanes 0-15 row M, K 0..15; lanes 16-31 K 16..31
  const int arow = mbase + (lane & 15);
  const int aoff = (lane < 16) ? 0 : 16;
  const _Float16* aptr = wh + (size_t)arow * KPAD + aoff;
  __builtin_prefetch(aptr, 0, 3);                          // global_prefetch_b8

  // B fragment, same contiguous k order as A: lane n<16 -> column N=n, K=0..15; lane n>=16 -> N=n-16, K=16..31
  const int fcol = lane & 15;
  const int boff = (lane < 16) ? 0 : 16;
  const _Float16* bptr = xs + fcol * HOP + boff;

  v8f c = {};
  for (int kk = 0; kk < KPAD; kk += 32) {
    union { uint4 q[2]; v16h v; } au;
    au.q[0] = *(const uint4*)(aptr + kk);                  // K = kk+aoff+0..7
    au.q[1] = *(const uint4*)(aptr + kk + 8);              // K = kk+aoff+8..15

    union { uint2 d[4]; v16h v; } bu;                      // 16 consecutive K halves
    const _Float16* bp = bptr + kk;                        // 8B-aligned (stride 200B)
    bu.d[0] = *(const uint2*)(bp);
    bu.d[1] = *(const uint2*)(bp + 4);
    bu.d[2] = *(const uint2*)(bp + 8);
    bu.d[3] = *(const uint2*)(bp + 12);

    c = __builtin_amdgcn_wmma_f32_16x16x32_f16(
        /*neg_a=*/false, au.v, /*neg_b=*/false, bu.v,
        /*c_mod=*/(short)0, c, /*reuse_a=*/false, /*reuse_b=*/false);
  }

  // ---- store C (f32 16x16 layout: VGPR r -> M = r (lanes 0-15) / r+8 (lanes 16-31)) ----
  const int fout = fbase + fcol;
  if (fout < NFRAMES) {
    const int mrow = mbase + ((lane < 16) ? 0 : 8);
    float* op = out + (size_t)b * NCH * NFRAMES + fout;
    #pragma unroll
    for (int r = 0; r < 8; ++r) {
      int m = mrow + r;
      if (m < NCH) op[(size_t)m * NFRAMES] = c[r];
    }
  }
}

// ---------------- launcher ----------------
extern "C" void kernel_launch(void* const* d_in, const int* in_sizes, int n_in,
                              void* d_out, int out_size, void* d_ws, size_t ws_size,
                              hipStream_t stream) {
  const float* x = (const float*)d_in[0];        // [32, 480000]
  const float* w = (const float*)d_in[1];        // [514, 1, 400]
  float* out = (float*)d_out;                    // [32, 514, 4803]

  _Float16* wh  = (_Float16*)d_ws;                               // 640*416*2   = 532,480 B
  _Float16* xph = (_Float16*)((char*)d_ws + (size_t)MPAD * KPAD * 2);  // 32*481920*2 ~ 30.8 MB

  prep_weight<<<(MPAD * KPAD) / 256, 256, 0, stream>>>(w, wh);
  prep_x<<<(int)(((size_t)NBATCH * XPLEN / 8 + 255) / 256), 256, 0, stream>>>(x, xph);

  dim3 grid(301 /*frame tiles*/, 5 /*8-wave channel groups*/, NBATCH);
  stft_wmma<<<grid, 256, 0, stream>>>(wh, xph, out);
}
